// DIFMultiHeadAttentionX_22557168239036
// MI455X (gfx1250) — matrix operand order repaired and checked by reading, not compiled, not run on hardware
//
// DIF multi-head attention for MI455X (gfx1250), wave32, bf16 WMMA pipeline.
// Workspace: 8 * 14,680,064 bf16 elements (~224 MiB) for Q/K/QP/KP/V/AQ/AK/CTX.
#include <hip/hip_runtime.h>
#include <hip/hip_bf16.h>

#define B_   1024
#define S_   100
#define SP_  112      // S padded to multiple of 16 for WMMA tiles
#define HID_ 128
#define NH_  2
#define DH_  64
#define AH_  64
#define DA_  32
#define F_   2

typedef __attribute__((ext_vector_type(16))) __bf16 v16bf;
typedef __attribute__((ext_vector_type(8)))  __bf16 v8bf;
typedef __attribute__((ext_vector_type(8)))  float  v8f;

static constexpr size_t N1      = (size_t)B_ * NH_ * SP_ * DH_;  // 14,680,064
static constexpr size_t OFF_Q   = 0 * N1;
static constexpr size_t OFF_K   = 1 * N1;
static constexpr size_t OFF_QP  = 2 * N1;
static constexpr size_t OFF_KP  = 3 * N1;
static constexpr size_t OFF_V   = 4 * N1;
static constexpr size_t OFF_AQ  = 5 * N1;   // F*B*NH*SP*DA == N1
static constexpr size_t OFF_AK  = 6 * N1;
static constexpr size_t OFF_CTX = 7 * N1;   // B*SP*HID == N1

// ---------------- WMMA helpers (CDNA5 16x16x32 bf16, f32 accum) ----------------

__device__ __forceinline__ v8f wmma_bf16(v16bf a, v16bf b, v8f c) {
  return __builtin_amdgcn_wmma_f32_16x16x32_bf16(false, a, false, b, (short)0, c,
                                                 false, false);
}

__device__ __forceinline__ v16bf frag_pack(const __bf16* p0, const __bf16* p1) {
  v8bf lo = *(const v8bf*)p0;   // ds_load_b128
  v8bf hi = *(const v8bf*)p1;   // ds_load_b128
  v16bf r;
#pragma unroll
  for (int e = 0; e < 8; ++e) { r[e] = lo[e]; r[e + 8] = hi[e]; }
  return r;
}

// A-matrix 16x32 bf16 fragment from row-major [M x K] (ISA 7.12.2 layout):
// lane L<16: row=mBase+L, K = kBase+{0..7,16..23}; L>=16: row=mBase+L-16, K=kBase+{8..15,24..31}
__device__ __forceinline__ v16bf frag_a(const __bf16* base, int lda, int mBase, int kBase) {
  int lane = threadIdx.x & 31;
  int hi   = (lane >> 4) & 1;
  const __bf16* p = base + (size_t)(mBase + (lane & 15)) * lda + kBase;
  return frag_pack(p + (hi ? 8 : 0), p + (hi ? 24 : 16));
}

// B-matrix 32x16 bf16 fragment; B stored N-major [N x K] so loads are contiguous:
// lane L: col = nBase + (L&15), K = kBase + {0..7,16..23} (L<16) or {8..15,24..31} (L>=16)
__device__ __forceinline__ v16bf frag_bT(const __bf16* base, int ldk, int kBase, int nBase) {
  int lane = threadIdx.x & 31;
  int hi   = (lane >> 4) & 1;
  const __bf16* p = base + (size_t)(nBase + (lane & 15)) * ldk + kBase;
  return frag_pack(p + (hi ? 8 : 0), p + (hi ? 24 : 16));
}

// ---------------- Kernel 1: item / positional projections ----------------
// One block per batch b. Out = [112,128] = X(bf16) @ W(bf16) + bias, split heads on store.

__global__ __launch_bounds__(256) void proj_item_kernel(
    const float* __restrict__ X, const float* __restrict__ P,
    const float* __restrict__ Wq, const float* __restrict__ bq,
    const float* __restrict__ Wk, const float* __restrict__ bk,
    const float* __restrict__ Wv, const float* __restrict__ bv,
    const float* __restrict__ Wqp, const float* __restrict__ bqp,
    const float* __restrict__ Wkp, const float* __restrict__ bkp,
    __bf16* __restrict__ ws) {
  extern __shared__ char smem[];
  __bf16* sX  = (__bf16*)smem;            // [112][128]
  __bf16* sP  = sX + SP_ * HID_;          // [112][128]
  __bf16* sWT = sP + SP_ * HID_;          // [N=128][K=128] (W transposed to N-major)
  float*  sBias = (float*)(sWT + HID_ * HID_);

  const int tid = threadIdx.x, lane = tid & 31, wave = tid >> 5;
  const int b = blockIdx.x;
  const float* Xg = X + (size_t)b * S_ * HID_;
  const float* Pg = P + (size_t)b * S_ * HID_;
  for (int i = tid; i < SP_ * HID_; i += 256) {
    int r = i >> 7, c = i & 127;
    float xv = (r < S_) ? Xg[r * HID_ + c] : 0.f;
    float pv = (r < S_) ? Pg[r * HID_ + c] : 0.f;
    sX[i] = (__bf16)xv;
    sP[i] = (__bf16)pv;
  }

  const float* Ws[5]   = {Wq, Wk, Wv, Wqp, Wkp};
  const float* Bs[5]   = {bq, bk, bv, bqp, bkp};
  const size_t Offs[5] = {OFF_Q, OFF_K, OFF_V, OFF_QP, OFF_KP};

  for (int p = 0; p < 5; ++p) {
    __syncthreads();
    if (p < 4) __builtin_prefetch(Ws[p + 1], 0, 0);  // global_prefetch_b8
    const float* W = Ws[p];
    for (int i = tid; i < HID_ * HID_; i += 256) {
      int n = i >> 7, k = i & 127;
      sWT[i] = (__bf16)W[k * HID_ + n];
    }
    if (tid < HID_) sBias[tid] = Bs[p][tid];
    __syncthreads();

    const __bf16* Asrc = (p >= 3) ? sP : sX;
    __bf16* dst = ws + Offs[p] + (size_t)b * NH_ * SP_ * DH_;
    const int nBase = wave * 16;  // 8 waves cover 128 output cols
    for (int mt = 0; mt < 7; ++mt) {
      v8f acc = {0.f, 0.f, 0.f, 0.f, 0.f, 0.f, 0.f, 0.f};
#pragma unroll
      for (int kk = 0; kk < 4; ++kk)
        acc = wmma_bf16(frag_a(Asrc, HID_, mt * 16, kk * 32),
                        frag_bT(sWT, HID_, kk * 32, nBase), acc);
      int col = nBase + (lane & 15), hi = lane >> 4;
      int head = col >> 6, d = col & 63;
      float bias = sBias[col];
#pragma unroll
      for (int r = 0; r < 8; ++r) {
        int row = mt * 16 + hi * 8 + r;
        float v = (row < S_) ? acc[r] + bias : 0.f;
        dst[((size_t)head * SP_ + row) * DH_ + d] = (__bf16)v;
      }
    }
  }
}

// ---------------- Kernel 2: attribute projections ----------------
// One block per (f,b). Out = [112,64] = attr(bf16) @ W_attr(bf16) + bias.

__global__ __launch_bounds__(256) void proj_attr_kernel(
    const float* __restrict__ ATTR,
    const float* __restrict__ Wq_attr, const float* __restrict__ bq_attr,
    const float* __restrict__ Wk_attr, const float* __restrict__ bk_attr,
    __bf16* __restrict__ ws) {
  extern __shared__ char smem[];
  __bf16* sA  = (__bf16*)smem;            // [112][64]
  __bf16* sWT = sA + SP_ * AH_;           // [N=64][K=64]
  float*  sBias = (float*)(sWT + AH_ * AH_);

  const int tid = threadIdx.x, lane = tid & 31, wave = tid >> 5;
  const int f = blockIdx.x >> 10, b = blockIdx.x & 1023;
  const float* Ag = ATTR + ((size_t)f * B_ + b) * S_ * AH_;
  for (int i = tid; i < SP_ * AH_; i += 256) {
    int r = i >> 6, c = i & 63;
    sA[i] = (__bf16)((r < S_) ? Ag[r * AH_ + c] : 0.f);
  }

  for (int p = 0; p < 2; ++p) {
    __syncthreads();
    const float* W  = (p ? Wk_attr : Wq_attr) + (size_t)f * AH_ * AH_;
    const float* bb = (p ? bk_attr : bq_attr) + (size_t)f * AH_;
    for (int i = tid; i < AH_ * AH_; i += 256) {
      int n = i >> 6, k = i & 63;
      sWT[i] = (__bf16)W[k * AH_ + n];
    }
    if (tid < AH_) sBias[tid] = bb[tid];
    __syncthreads();

    __bf16* dst = ws + (p ? OFF_AK : OFF_AQ) + ((size_t)f * B_ + b) * NH_ * SP_ * DA_;
    for (int t = wave; t < 28; t += 8) {  // 7 row-tiles x 4 col-tiles
      int mt = t >> 2, nt = t & 3;
      v8f acc = {0.f, 0.f, 0.f, 0.f, 0.f, 0.f, 0.f, 0.f};
#pragma unroll
      for (int kk = 0; kk < 2; ++kk)
        acc = wmma_bf16(frag_a(sA, AH_, mt * 16, kk * 32),
                        frag_bT(sWT, AH_, kk * 32, nt * 16), acc);
      int col = nt * 16 + (lane & 15), hi = lane >> 4;
      int head = col >> 5, d = col & 31;
      float bias = sBias[col];
#pragma unroll
      for (int r = 0; r < 8; ++r) {
        int row = mt * 16 + hi * 8 + r;
        float v = (row < S_) ? acc[r] + bias : 0.f;
        dst[((size_t)head * SP_ + row) * DA_ + d] = (__bf16)v;
      }
    }
  }
}

// ---------------- Kernel 3: score streams + gated fusion + attention ----------------
// One block per (b,h). ~200 KB LDS (CDNA5 WGP has 320 KB).

__global__ __launch_bounds__(256) void attn_kernel(
    __bf16* __restrict__ ws, const float* __restrict__ mask,
    const float* __restrict__ W1, const float* __restrict__ b1,
    const float* __restrict__ W2) {
  extern __shared__ char smem[];
  __bf16* sW1T = (__bf16*)smem;              // [N=112][K=128] fuse_W1^T, zero padded
  __bf16* sSc  = sW1T + SP_ * HID_;          // 4 streams: [448][112] bf16 rows
  __bf16* sPr  = sSc + 4 * SP_ * SP_;        // probs [112][128] (K-padded to 128)
  __bf16* sVT  = sPr + SP_ * HID_;           // V^T: [N=64][K=128], rows>=112 zero
  __bf16* sA   = sVT + DH_ * HID_;           // A operand [112][<=64]
  __bf16* sB   = sA + SP_ * DH_;             // B operand (K row-major == N-major) [112][<=64]
  float*  sEn  = (float*)(sB + SP_ * DH_);   // energies [4][112]
  float*  sB1  = sEn + 4 * SP_;              // fuse_b1 padded
  float*  sW2v = sB1 + SP_;                  // fuse_W2 padded

  const int tid = threadIdx.x, lane = tid & 31, wave = tid >> 5;
  const int b = blockIdx.x >> 1, h = blockIdx.x & 1;

  // fuse params -> LDS (zero padded so WMMA K-overrun contributes 0)
  for (int i = tid; i < SP_ * HID_; i += 256) {
    int n = i >> 7, k = i & 127;
    sW1T[i] = (__bf16)((n < S_ && k < S_) ? W1[k * S_ + n] : 0.f);
  }
  for (int i = tid; i < SP_; i += 256) {
    sB1[i]  = (i < S_) ? b1[i] : 0.f;
    sW2v[i] = (i < S_) ? W2[i] : 0.f;
  }
  // V^T into LDS
  {
    const __bf16* Vg = ws + OFF_V + (size_t)(b * NH_ + h) * SP_ * DH_;
    for (int i = tid; i < DH_ * HID_; i += 256) {
      int n = i >> 7, k = i & 127;
      sVT[i] = (k < SP_) ? Vg[k * DH_ + n] : (__bf16)0.f;
    }
  }

  // ---- 4 score streams: attr0, attr1, item, pos (reference stacking order) ----
  for (int st = 0; st < 4; ++st) {
    __syncthreads();  // previous stream's WMMAs done before overwriting sA/sB
    int lda, kT;
    if (st < 2) {
      const __bf16* Ag = ws + OFF_AQ + (((size_t)st * B_ + b) * NH_ + h) * SP_ * DA_;
      const __bf16* Bg = ws + OFF_AK + (((size_t)st * B_ + b) * NH_ + h) * SP_ * DA_;
      const uint4* ga = (const uint4*)Ag; const uint4* gb = (const uint4*)Bg;
      uint4* pa = (uint4*)sA; uint4* pb = (uint4*)sB;
      for (int i = tid; i < (SP_ * DA_) / 8; i += 256) { pa[i] = ga[i]; pb[i] = gb[i]; }
      lda = DA_; kT = 1;
    } else {
      const __bf16* Ag = ws + (st == 2 ? OFF_Q : OFF_QP) + (size_t)(b * NH_ + h) * SP_ * DH_;
      const __bf16* Bg = ws + (st == 2 ? OFF_K : OFF_KP) + (size_t)(b * NH_ + h) * SP_ * DH_;
      const uint4* ga = (const uint4*)Ag; const uint4* gb = (const uint4*)Bg;
      uint4* pa = (uint4*)sA; uint4* pb = (uint4*)sB;
      for (int i = tid; i < (SP_ * DH_) / 8; i += 256) { pa[i] = ga[i]; pb[i] = gb[i]; }
      lda = DH_; kT = 2;
    }
    __syncthreads();
    for (int t = wave; t < 49; t += 8) {  // [112,112] scores, 7x7 tiles
      int mt = t / 7, nt = t % 7;
      v8f acc = {0.f, 0.f, 0.f, 0.f, 0.f, 0.f, 0.f, 0.f};
      for (int kk = 0; kk < kT; ++kk)
        acc = wmma_bf16(frag_a(sA, lda, mt * 16, kk * 32),
                        frag_bT(sB, lda, kk * 32, nt * 16), acc);
      int col = nt * 16 + (lane & 15), hi = lane >> 4;
#pragma unroll
      for (int r = 0; r < 8; ++r) {
        int row = mt * 16 + hi * 8 + r;
        sSc[(size_t)st * SP_ * SP_ + row * SP_ + col] = (__bf16)acc[r];
      }
    }
  }

  // ---- fusion energies: E = relu(Rows @ W1 + b1) . W2, Rows = [448][112] bf16 ----
  __syncthreads();
  for (int i = tid; i < 4 * SP_; i += 256) sEn[i] = 0.f;
  __syncthreads();
  for (int mt = wave; mt < 28; mt += 8) {   // 448 rows = 28 row-tiles
    float part[8] = {0.f, 0.f, 0.f, 0.f, 0.f, 0.f, 0.f, 0.f};
    for (int nt = 0; nt < 7; ++nt) {
      v8f acc = {0.f, 0.f, 0.f, 0.f, 0.f, 0.f, 0.f, 0.f};
#pragma unroll
      for (int kk = 0; kk < 4; ++kk)  // K padded to 128; W1 rows>=100 are zero
        acc = wmma_bf16(frag_a(sSc, SP_, mt * 16, kk * 32),
                        frag_bT(sW1T, HID_, kk * 32, nt * 16), acc);
      int col = nt * 16 + (lane & 15);
      float w2 = sW2v[col], bb = sB1[col];
#pragma unroll
      for (int r = 0; r < 8; ++r) {
        float e = acc[r] + bb;
        part[r] += (e > 0.f ? e : 0.f) * w2;
      }
    }
    int hi = lane >> 4;
#pragma unroll
    for (int r = 0; r < 8; ++r)
      atomicAdd(&sEn[mt * 16 + hi * 8 + r], part[r]);  // ds_add_f32
  }
  __syncthreads();

  // ---- gate softmax over streams, fuse, mask, key softmax -> probs (bf16) ----
  const float* maskB = mask + (size_t)b * S_ * S_;
  for (int s = wave; s < S_; s += 8) {
    float e0 = sEn[0 * SP_ + s], e1 = sEn[1 * SP_ + s];
    float e2 = sEn[2 * SP_ + s], e3 = sEn[3 * SP_ + s];
    float em = fmaxf(fmaxf(e0, e1), fmaxf(e2, e3));
    float g0 = __expf(e0 - em), g1 = __expf(e1 - em);
    float g2 = __expf(e2 - em), g3 = __expf(e3 - em);
    float gs = 1.f / (g0 + g1 + g2 + g3);
    g0 *= gs; g1 *= gs; g2 *= gs; g3 *= gs;
    const __bf16* r0 = &sSc[0 * SP_ * SP_ + s * SP_];
    const __bf16* r1 = &sSc[1 * SP_ * SP_ + s * SP_];
    const __bf16* r2 = &sSc[2 * SP_ * SP_ + s * SP_];
    const __bf16* r3 = &sSc[3 * SP_ * SP_ + s * SP_];
    float vals[4]; float rmax = -3.0e38f;
#pragma unroll
    for (int j = 0; j < 4; ++j) {
      int k = lane + j * 32;
      float fv = -3.0e38f;
      if (k < S_) {
        float x = g0 * (float)r0[k] + g1 * (float)r1[k] +
                  g2 * (float)r2[k] + g3 * (float)r3[k];
        fv = x * 0.125f + maskB[s * S_ + k];   // 1/sqrt(DH) = 0.125
      }
      vals[j] = fv; rmax = fmaxf(rmax, fv);
    }
    for (int off = 16; off; off >>= 1) rmax = fmaxf(rmax, __shfl_xor(rmax, off, 32));
    float rsum = 0.f;
#pragma unroll
    for (int j = 0; j < 4; ++j) {
      int k = lane + j * 32;
      float pv = (k < S_) ? __expf(vals[j] - rmax) : 0.f;
      vals[j] = pv; rsum += pv;
    }
    for (int off = 16; off; off >>= 1) rsum += __shfl_xor(rsum, off, 32);
    float inv = 1.f / rsum;
#pragma unroll
    for (int j = 0; j < 4; ++j) {
      int k = lane + j * 32;
      sPr[s * HID_ + k] = (__bf16)((k < S_) ? vals[j] * inv : 0.f);
    }
  }
  for (int i = tid; i < (SP_ - S_) * HID_; i += 256)  // zero padded query rows
    sPr[S_ * HID_ + i] = (__bf16)0.f;
  __syncthreads();

  // ---- ctx = probs[112,128pad] @ V[128pad,64] ----
  __bf16* ctxW = ws + OFF_CTX;
  for (int t = wave; t < 28; t += 8) {  // 7 x 4 tiles
    int mt = t >> 2, nt = t & 3;
    v8f acc = {0.f, 0.f, 0.f, 0.f, 0.f, 0.f, 0.f, 0.f};
#pragma unroll
    for (int kk = 0; kk < 4; ++kk)
      acc = wmma_bf16(frag_a(sPr, HID_, mt * 16, kk * 32),
                      frag_bT(sVT, HID_, kk * 32, nt * 16), acc);
    int col = nt * 16 + (lane & 15), hi = lane >> 4;
#pragma unroll
    for (int r = 0; r < 8; ++r) {
      int row = mt * 16 + hi * 8 + r;
      ctxW[((size_t)b * SP_ + row) * HID_ + h * DH_ + col] = (__bf16)acc[r];
    }
  }
}

// ---------------- Kernel 4: output dense + residual + LayerNorm ----------------

__global__ __launch_bounds__(256) void out_kernel(
    const float* __restrict__ X, const float* __restrict__ Wd,
    const float* __restrict__ bd, const float* __restrict__ ln_g,
    const float* __restrict__ ln_b, const __bf16* __restrict__ ws,
    float* __restrict__ out) {
  extern __shared__ char smem[];
  __bf16* sC  = (__bf16*)smem;              // ctx [112][128] bf16
  __bf16* sWT = sC + SP_ * HID_;            // Wd^T [N=128][K=128]
  float*  sH  = (float*)(sWT + HID_ * HID_); // h [112][128] f32
  float*  sBd = sH + SP_ * HID_;
  float*  sG  = sBd + HID_;
  float*  sBt = sG + HID_;

  const int tid = threadIdx.x, lane = tid & 31, wave = tid >> 5;
  const int b = blockIdx.x;
  {
    const uint4* g = (const uint4*)(ws + OFF_CTX + (size_t)b * SP_ * HID_);
    uint4* s = (uint4*)sC;
    for (int i = tid; i < (SP_ * HID_) / 8; i += 256) s[i] = g[i];
  }
  for (int i = tid; i < HID_ * HID_; i += 256) {
    int n = i >> 7, k = i & 127;
    sWT[i] = (__bf16)Wd[k * HID_ + n];
  }
  if (tid < HID_) { sBd[tid] = bd[tid]; sG[tid] = ln_g[tid]; sBt[tid] = ln_b[tid]; }
  __syncthreads();

  const float* Xg = X + (size_t)b * S_ * HID_;
  const int nBase = wave * 16;
  for (int mt = 0; mt < 7; ++mt) {
    v8f acc = {0.f, 0.f, 0.f, 0.f, 0.f, 0.f, 0.f, 0.f};
#pragma unroll
    for (int kk = 0; kk < 4; ++kk)
      acc = wmma_bf16(frag_a(sC, HID_, mt * 16, kk * 32),
                      frag_bT(sWT, HID_, kk * 32, nBase), acc);
    int col = nBase + (lane & 15), hi = lane >> 4;
#pragma unroll
    for (int r = 0; r < 8; ++r) {
      int row = mt * 16 + hi * 8 + r;
      if (row < S_)
        sH[row * HID_ + col] = acc[r] + sBd[col] + Xg[row * HID_ + col];
    }
  }
  __syncthreads();

  float* outB = out + (size_t)b * S_ * HID_;
  for (int s = wave; s < S_; s += 8) {
    int c = lane * 4;
    float x0 = sH[s * HID_ + c + 0], x1 = sH[s * HID_ + c + 1];
    float x2 = sH[s * HID_ + c + 2], x3 = sH[s * HID_ + c + 3];
    float sum = x0 + x1 + x2 + x3;
    for (int off = 16; off; off >>= 1) sum += __shfl_xor(sum, off, 32);
    float mu = sum * (1.f / HID_);
    float d0 = x0 - mu, d1 = x1 - mu, d2 = x2 - mu, d3 = x3 - mu;
    float ss = d0 * d0 + d1 * d1 + d2 * d2 + d3 * d3;
    for (int off = 16; off; off >>= 1) ss += __shfl_xor(ss, off, 32);
    float inv = rsqrtf(ss * (1.f / HID_) + 1e-12f);
    outB[s * HID_ + c + 0] = d0 * inv * sG[c + 0] + sBt[c + 0];
    outB[s * HID_ + c + 1] = d1 * inv * sG[c + 1] + sBt[c + 1];
    outB[s * HID_ + c + 2] = d2 * inv * sG[c + 2] + sBt[c + 2];
    outB[s * HID_ + c + 3] = d3 * inv * sG[c + 3] + sBt[c + 3];
  }
}

// ---------------- host launcher ----------------

extern "C" void kernel_launch(void* const* d_in, const int* in_sizes, int n_in,
                              void* d_out, int out_size, void* d_ws, size_t ws_size,
                              hipStream_t stream) {
  (void)in_sizes; (void)n_in; (void)out_size; (void)ws_size;
  const float* X    = (const float*)d_in[0];
  const float* ATTR = (const float*)d_in[1];
  const float* P    = (const float*)d_in[2];
  const float* MASK = (const float*)d_in[3];
  const float* Wq  = (const float*)d_in[4],  *bq  = (const float*)d_in[5];
  const float* Wk  = (const float*)d_in[6],  *bk  = (const float*)d_in[7];
  const float* Wv  = (const float*)d_in[8],  *bv  = (const float*)d_in[9];
  const float* Wqp = (const float*)d_in[10], *bqp = (const float*)d_in[11];
  const float* Wkp = (const float*)d_in[12], *bkp = (const float*)d_in[13];
  const float* Wq_attr = (const float*)d_in[14], *bq_attr = (const float*)d_in[15];
  const float* Wk_attr = (const float*)d_in[16], *bk_attr = (const float*)d_in[17];
  const float* fW1 = (const float*)d_in[18], *fb1 = (const float*)d_in[19];
  const float* fW2 = (const float*)d_in[20];  // fuse_b2 (d_in[21]) cancels in softmax
  const float* Wd  = (const float*)d_in[22], *bd  = (const float*)d_in[23];
  const float* lng = (const float*)d_in[24], *lnb = (const float*)d_in[25];
  __bf16* ws = (__bf16*)d_ws;
  float* out = (float*)d_out;

  constexpr int SMEM1 = (2 * SP_ * HID_ + HID_ * HID_) * 2 + HID_ * 4;            // 90,624
  constexpr int SMEM2 = (SP_ * AH_ + AH_ * AH_) * 2 + AH_ * 4;                    // 22,784
  constexpr int SMEM3 = (SP_ * HID_ + 4 * SP_ * SP_ + SP_ * HID_ + DH_ * HID_ +
                         2 * SP_ * DH_) * 2 + (6 * SP_) * 4;                      // 205,440
  constexpr int SMEM4 = (SP_ * HID_ + HID_ * HID_) * 2 + SP_ * HID_ * 4 + 3 * HID_ * 4; // 120,320

  (void)hipFuncSetAttribute((const void*)proj_item_kernel,
                            hipFuncAttributeMaxDynamicSharedMemorySize, SMEM1);
  (void)hipFuncSetAttribute((const void*)proj_attr_kernel,
                            hipFuncAttributeMaxDynamicSharedMemorySize, SMEM2);
  (void)hipFuncSetAttribute((const void*)attn_kernel,
                            hipFuncAttributeMaxDynamicSharedMemorySize, SMEM3);
  (void)hipFuncSetAttribute((const void*)out_kernel,
                            hipFuncAttributeMaxDynamicSharedMemorySize, SMEM4);

  proj_item_kernel<<<dim3(B_), dim3(256), SMEM1, stream>>>(
      X, P, Wq, bq, Wk, bk, Wv, bv, Wqp, bqp, Wkp, bkp, ws);
  proj_attr_kernel<<<dim3(F_ * B_), dim3(256), SMEM2, stream>>>(
      ATTR, Wq_attr, bq_attr, Wk_attr, bk_attr, ws);
  attn_kernel<<<dim3(B_ * NH_), dim3(256), SMEM3, stream>>>(ws, MASK, fW1, fb1, fW2);
  out_kernel<<<dim3(B_), dim3(256), SMEM4, stream>>>(X, Wd, bd, lng, lnb, ws, out);
}
